// Attention_33827162423573
// MI455X (gfx1250) — matrix operand order repaired and checked by reading, not compiled, not run on hardware
//
#include <hip/hip_runtime.h>
#include <hip/hip_bf16.h>

typedef _Float16 h16;
typedef __attribute__((ext_vector_type(16))) _Float16 v16h;
typedef __attribute__((ext_vector_type(8)))  _Float16 v8h;
typedef __attribute__((ext_vector_type(8)))  float    v8f;
typedef unsigned int u32x4 __attribute__((ext_vector_type(4)));
typedef int          i32x4 __attribute__((ext_vector_type(4)));
typedef int          i32x8 __attribute__((ext_vector_type(8)));

// ---------------------------------------------------------------------------
// WMMA helpers (CDNA5 wave32 layouts, ISA 05_wmma §7.12.2):
//  A (16x32 f16, M x K): lane L: m = L%16, g = L/16; code's order j -> K = g*16 + j (not ISA table)
//                        => two 16B chunks.
//  B (32x16 f16, stored as N x K row-major = "NT" operand): lane L: n = L%16,
//                        g = L/16; code's order j -> K = g*16 + j, as A => one 32B chunk.
//  C/D (16x16 f32): VGPR r, lane L: row m = r + 8*(L/16), col n = L%16.
// ---------------------------------------------------------------------------
__device__ __forceinline__ v16h load_a_frag(const h16* row, int kbase, int g) {
  const v8h lo = *(const v8h*)(row + kbase + g * 16);
  const v8h hi = *(const v8h*)(row + kbase + g * 16 + 8);
  return __builtin_shufflevector(lo, hi, 0,1,2,3,4,5,6,7,8,9,10,11,12,13,14,15);
}
__device__ __forceinline__ v16h load_b_frag(const h16* p) {
  const v8h lo = *(const v8h*)(p);
  const v8h hi = *(const v8h*)(p + 8);
  return __builtin_shufflevector(lo, hi, 0,1,2,3,4,5,6,7,8,9,10,11,12,13,14,15);
}
__device__ __forceinline__ v8f wmma_f16(v16h a, v16h b, v8f c) {
  return __builtin_amdgcn_wmma_f32_16x16x32_f16(false, a, false, b, (short)0, c,
                                                false, false);
}

// Async global->LDS copy (ISA 08 §4): per lane LDS[vdst] <- MEM[vaddr], 16B.
// LDS byte offset = low 32 bits of the generic shared-memory address.
__device__ __forceinline__ void async_copy_b128(void* lds, const void* gptr) {
  unsigned loff = (unsigned)(uintptr_t)lds;
  asm volatile("global_load_async_to_lds_b128 %0, %1, off"
               :: "v"(loff), "v"(gptr) : "memory");
}
__device__ __forceinline__ void wait_asynccnt0() {
  asm volatile("s_wait_asynccnt 0" ::: "memory");
}

// Tensor Data Mover: DMA a 64x64 2-byte tile (row-major, stride 64) into LDS
// with 4 DWORDs of padding after every 32 DWORDs => LDS row stride 72 halves.
// D# per ISA 08 §8.3-8.5: count=1, type=2(image), data_size=1(2B),
// pad_enable=1, pad_interval=4 (32 DWORDs), pad_amount=3 (4 DWORDs),
// tensor_dim0/1 = tile_dim0/1 = 64, tensor_dim0_stride = 64, wg_mask=0.
__device__ __forceinline__ void tdm_load_64x64_padded(void* lds, const void* gsrc) {
  unsigned lo = (unsigned)(uintptr_t)lds;
  unsigned long long ga = (unsigned long long)(uintptr_t)gsrc;
  u32x4 g0 = { 1u,                                               // count=1
               lo,                                               // lds_addr
               (unsigned)ga,                                     // global_addr lo
               ((unsigned)(ga >> 32) & 0x01FFFFFFu) | (2u << 30) };  // type=2
  i32x8 g1 = { (int)((1u << 16) | (1u << 20) | (4u << 22) | (3u << 25)),
               (int)(64u << 16),   // tensor_dim0[15:0] in bits[63:48]
               (int)(64u << 16),   // tensor_dim1[15:0] in bits[95:80]
               (int)(64u << 16),   // tile_dim0 in bits[127:112]
               64,                 // tile_dim1 in bits[143:128]
               64,                 // tensor_dim0_stride[31:0]
               0, 0 };
  i32x4 z4 = {0, 0, 0, 0};
#if __has_include(<hip/amd_detail/amd_gfx1250_TDM.h>)
  i32x8 z8 = {0, 0, 0, 0, 0, 0, 0, 0};
  __builtin_amdgcn_tensor_load_to_lds(g0, g1, z4, z4, z8, 0);
#else
  __builtin_amdgcn_tensor_load_to_lds(g0, g1, z4, z4, 0);
#endif
}

// ---------------------------------------------------------------------------
// Pack kernels
// ---------------------------------------------------------------------------
__global__ void cvt_f32_f16(const float* __restrict__ s, h16* __restrict__ d, int n) {
  for (int i = blockIdx.x * blockDim.x + threadIdx.x; i < n; i += gridDim.x * blockDim.x)
    d[i] = (h16)s[i];
}

// wr[oc][kh][kw][ic]  (oc<512: q_w ; oc in [512,1536): kv_w) ; K-contiguous ic
__global__ void pack_qkv_w(const float* __restrict__ q_w, const float* __restrict__ kv_w,
                           h16* __restrict__ wr) {
  const int n = 1536 * 9 * 512;
  for (int i = blockIdx.x * blockDim.x + threadIdx.x; i < n; i += gridDim.x * blockDim.x) {
    int oc = i / 4608, rem = i % 4608, tap = rem / 512, ic = rem % 512;
    float v = (oc < 512) ? q_w[(oc * 512 + ic) * 9 + tap]
                         : kv_w[((oc - 512) * 512 + ic) * 9 + tap];
    wr[i] = (h16)v;
  }
}

// ---------------------------------------------------------------------------
// Fused q/k/v 3x3 conv as implicit GEMM.
//  M = 1536 oc, N = B*H*W px; one WG = (b, rows h0..h0+1) x 128 oc;
//  K = 9 taps * 512 ic stepped as (tap, 32-ic chunk). 8 waves, each
//  16 oc x 128 px => acc[2][4]. Inner loop: kw outer, 4-row B-fragment
//  sliding reuse => 72 WMMA but only 48 B-fragment loads per ic-chunk.
// ---------------------------------------------------------------------------
#define CCELL 40  // halves per (row,col) cell in LDS: 80B, 16B aligned, bank-spread
__global__ __launch_bounds__(256) void qkv_conv_kernel(
    const h16* __restrict__ xh, const h16* __restrict__ wr,
    const float* __restrict__ q_b, const float* __restrict__ kv_b,
    h16* __restrict__ Q, h16* __restrict__ K, h16* __restrict__ V) {
  __shared__ h16 lds_x[4 * 66 * CCELL];  // [row 0..3][col 0..65][ic 0..31]
  const int b = blockIdx.x >> 5;
  const int h0 = (blockIdx.x & 31) * 2;
  const int oc0 = blockIdx.y * 128;
  const int tid = threadIdx.x, lane = tid & 31, wid = tid >> 5;
  const int n = lane & 15, g = lane >> 4;
  const int woc = oc0 + wid * 16;  // wave's 16-oc block; A-fragment row m == n

  // Zero the loop-invariant padding cells once (cols 0/65, out-of-image rows).
  for (int idx = tid; idx < 4 * 66 * 32; idx += 256) {
    int c = idx % 66, t = idx / 66, r = t & 3, ic = t >> 2;
    int hh = h0 + r - 1;
    if ((c == 0) | (c == 65) | ((unsigned)hh >= 64u))
      lds_x[(r * 66 + c) * CCELL + ic] = (h16)0.f;
  }

  v8f acc[2][4] = {};
  for (int ic0 = 0; ic0 < 512; ic0 += 32) {
    // Stage interior: 32 ic x 4 rows x 64 cols, 16B global loads, ic-fastest LDS.
    for (int idx = tid; idx < 4 * 32 * 8; idx += 256) {
      int j = idx & 7, t = idx >> 3, r = t & 3, ic = t >> 2;
      int hh = h0 + r - 1;
      if ((unsigned)hh < 64u) {
        const h16* src = &xh[(((size_t)b * 512 + ic0 + ic) * 64 + hh) * 64 + j * 8];
        v8h vv = *(const v8h*)src;
        if (ic0 + 32 < 512)  // next ic-chunk -> global_prefetch_b8
          __builtin_prefetch(src + (size_t)32 * 4096, 0, 1);
#pragma unroll
        for (int u = 0; u < 8; ++u)
          lds_x[(r * 66 + 1 + j * 8 + u) * CCELL + ic] = vv[u];
      }
    }
    __syncthreads();

    const h16* wbase = wr + (size_t)(woc + n) * 4608 + ic0;
#pragma unroll
    for (int kw = 0; kw < 3; ++kw) {
      v16h a0 = load_a_frag(wbase + (0 + kw) * 512, 0, g);  // kh = 0
      v16h a1 = load_a_frag(wbase + (3 + kw) * 512, 0, g);  // kh = 1
      v16h a2 = load_a_frag(wbase + (6 + kw) * 512, 0, g);  // kh = 2
#pragma unroll
      for (int t4 = 0; t4 < 4; ++t4) {
        const int col = t4 * 16 + n + kw;
        v16h r0 = load_b_frag(&lds_x[(0 * 66 + col) * CCELL + g * 16]);
        v16h r1 = load_b_frag(&lds_x[(1 * 66 + col) * CCELL + g * 16]);
        v16h r2 = load_b_frag(&lds_x[(2 * 66 + col) * CCELL + g * 16]);
        v16h r3 = load_b_frag(&lds_x[(3 * 66 + col) * CCELL + g * 16]);
        // B row (ro + kh); r1, r2 each reused by two WMMAs.
        acc[0][t4] = wmma_f16(a0, r0, acc[0][t4]);
        acc[1][t4] = wmma_f16(a0, r1, acc[1][t4]);
        acc[0][t4] = wmma_f16(a1, r1, acc[0][t4]);
        acc[1][t4] = wmma_f16(a1, r2, acc[1][t4]);
        acc[0][t4] = wmma_f16(a2, r2, acc[0][t4]);
        acc[1][t4] = wmma_f16(a2, r3, acc[1][t4]);
      }
    }
    __syncthreads();
  }

  // Epilogue: bias + f16 store into Q / K / V (natural [b][ch][h][w] layout).
#pragma unroll
  for (int r = 0; r < 8; ++r) {
    int ocg = woc + r + 8 * g;  // D row m = r + 8g
    float bias;
    h16* dst;
    int ch;
    if (ocg < 512)        { bias = q_b[ocg];        dst = Q; ch = ocg; }
    else if (ocg < 1024)  { bias = kv_b[ocg - 512]; dst = K; ch = ocg - 512; }
    else                  { bias = kv_b[ocg - 512]; dst = V; ch = ocg - 1024; }
#pragma unroll
    for (int ro = 0; ro < 2; ++ro) {
      size_t base = (((size_t)b * 512 + ch) * 64 + (h0 + ro)) * 64;
#pragma unroll
      for (int t4 = 0; t4 < 4; ++t4)
        dst[base + t4 * 16 + n] = (h16)(acc[ro][t4][r] + bias);
    }
  }
}

// ---------------------------------------------------------------------------
// Per-(b,ch) 64x64 row-attention: S = q k^T, softmax rows, O = P v.
// One WG (4 wave32) per channel. q staged by the Tensor Data Mover (single
// tensor_load_to_lds with hardware LDS padding), k staged via
// GLOBAL_LOAD_ASYNC_TO_LDS_B128, v transposed through registers into LDS for
// the NT second GEMM. LDS stride 72 halves = 144B (16B-aligned, bank-spread).
// Output gets the reference's (nh,c)->(c,nh) channel permutation.
// ---------------------------------------------------------------------------
#define ASTR 72
__global__ __launch_bounds__(128) void attn_kernel(
    const h16* __restrict__ Q, const h16* __restrict__ K,
    const h16* __restrict__ V, h16* __restrict__ AO) {
  __shared__ h16 qs[64 * ASTR], ks[64 * ASTR], vts[64 * ASTR], ps[64 * ASTR];
  const int ch = blockIdx.x & 511, b = blockIdx.x >> 9;
  const int tid = threadIdx.x, lane = tid & 31, wid = tid >> 5;
  const int n = lane & 15, g = lane >> 4;
  const size_t base = ((size_t)b * 512 + ch) * 4096;

  // q: one TDM descriptor issued by wave 0 (TENSORcnt-tracked DMA).
  if (wid == 0) tdm_load_64x64_padded(&qs[0], Q + base);

  // k: async-copy 512 16B chunks; LDS row stride 72 halves.
#pragma unroll
  for (int i = 0; i < 4; ++i) {
    int idx = tid + i * 128;             // chunk id 0..511
    int row = idx >> 3, cc = (idx & 7) * 8;
    async_copy_b128(&ks[row * ASTR + cc], K + base + idx * 8);
  }
  // v: load 16B, scatter transposed (vts[w][h]).
  for (int idx = tid; idx < 512; idx += 128) {
    int r = idx >> 3, c0 = (idx & 7) * 8;
    v8h vv = *(const v8h*)(V + base + (r << 6) + c0);
#pragma unroll
    for (int j = 0; j < 8; ++j) vts[(c0 + j) * ASTR + r] = vv[j];
  }
  wait_asynccnt0();
  if (wid == 0) __builtin_amdgcn_s_wait_tensorcnt(0);
  __syncthreads();

  const int m0 = wid * 16;  // this wave owns S/O rows m0..m0+15
  v8f acc[4] = {};
#pragma unroll
  for (int kk = 0; kk < 64; kk += 32) {
    v16h a = load_a_frag(&qs[(m0 + n) * ASTR], kk, g);
    v16h bf0 = load_b_frag(&ks[(0 * 16 + n) * ASTR + kk + g * 16]);
    v16h bf1 = load_b_frag(&ks[(1 * 16 + n) * ASTR + kk + g * 16]);
    v16h bf2 = load_b_frag(&ks[(2 * 16 + n) * ASTR + kk + g * 16]);
    v16h bf3 = load_b_frag(&ks[(3 * 16 + n) * ASTR + kk + g * 16]);
    acc[0] = wmma_f16(a, bf0, acc[0]);
    acc[1] = wmma_f16(a, bf1, acc[1]);
    acc[2] = wmma_f16(a, bf2, acc[2]);
    acc[3] = wmma_f16(a, bf3, acc[3]);
  }

  // Row softmax on D fragments: row m = m0 + r + 8g spans 16 lanes of a g-group.
#pragma unroll
  for (int r = 0; r < 8; ++r) {
    float mx = fmaxf(fmaxf(acc[0][r], acc[1][r]), fmaxf(acc[2][r], acc[3][r]));
    for (int off = 1; off < 16; off <<= 1) mx = fmaxf(mx, __shfl_xor(mx, off, 32));
    float s = 0.f;
#pragma unroll
    for (int t = 0; t < 4; ++t) {
      float e = __expf(acc[t][r] - mx);
      acc[t][r] = e;
      s += e;
    }
    for (int off = 1; off < 16; off <<= 1) s += __shfl_xor(s, off, 32);
    float inv = 1.0f / s;
    int row = m0 + r + 8 * g;
#pragma unroll
    for (int t = 0; t < 4; ++t)
      ps[row * ASTR + t * 16 + n] = (h16)(acc[t][r] * inv);
  }
  __syncthreads();

  v8f o[4] = {};
#pragma unroll
  for (int kk = 0; kk < 64; kk += 32) {
    v16h a = load_a_frag(&ps[(m0 + n) * ASTR], kk, g);
    v16h bf0 = load_b_frag(&vts[(0 * 16 + n) * ASTR + kk + g * 16]);
    v16h bf1 = load_b_frag(&vts[(1 * 16 + n) * ASTR + kk + g * 16]);
    v16h bf2 = load_b_frag(&vts[(2 * 16 + n) * ASTR + kk + g * 16]);
    v16h bf3 = load_b_frag(&vts[(3 * 16 + n) * ASTR + kk + g * 16]);
    o[0] = wmma_f16(a, bf0, o[0]);
    o[1] = wmma_f16(a, bf1, o[1]);
    o[2] = wmma_f16(a, bf2, o[2]);
    o[3] = wmma_f16(a, bf3, o[3]);
  }

  const int chp = (ch & 31) * 16 + (ch >> 5);  // (nh,c) -> (c,nh) permutation
  const size_t obase = ((size_t)b * 512 + chp) * 4096;
#pragma unroll
  for (int r = 0; r < 8; ++r) {
    int row = m0 + r + 8 * g;
#pragma unroll
    for (int t = 0; t < 4; ++t)
      AO[obase + row * 64 + t * 16 + n] = (h16)o[t][r];
  }
}

// ---------------------------------------------------------------------------
// 1x1 projection conv: GEMM M=512 oc x K=512 ic x N=B*H*W. B tiles transposed
// through LDS so the ic (K) axis is contiguous per pixel. f32 output + bias.
// ---------------------------------------------------------------------------
#define PCELL 40
__global__ __launch_bounds__(256) void proj_kernel(
    const h16* __restrict__ AO, const h16* __restrict__ wp,
    const float* __restrict__ proj_b, float* __restrict__ out) {
  __shared__ h16 lds_b[64 * PCELL];  // [pixel 0..63][ic 0..31]
  const int b = blockIdx.x >> 6, h = blockIdx.x & 63;
  const int oc0 = blockIdx.y * 128;
  const int tid = threadIdx.x, lane = tid & 31, wid = tid >> 5;
  const int n = lane & 15, g = lane >> 4;
  const int woc = oc0 + wid * 16;

  v8f acc[4] = {};
  for (int ic0 = 0; ic0 < 512; ic0 += 32) {
    {
      int ic = tid >> 3, c0 = (tid & 7) * 8;  // 256 chunks of 8 halves
      v8h vv = *(const v8h*)(AO + (((size_t)b * 512 + ic0 + ic) * 64 + h) * 64 + c0);
#pragma unroll
      for (int j = 0; j < 8; ++j) lds_b[(c0 + j) * PCELL + ic] = vv[j];
    }
    __syncthreads();
    v16h a = load_a_frag(wp + (size_t)(woc + n) * 512 + ic0, 0, g);
    v16h bf0 = load_b_frag(&lds_b[(0 * 16 + n) * PCELL + g * 16]);
    v16h bf1 = load_b_frag(&lds_b[(1 * 16 + n) * PCELL + g * 16]);
    v16h bf2 = load_b_frag(&lds_b[(2 * 16 + n) * PCELL + g * 16]);
    v16h bf3 = load_b_frag(&lds_b[(3 * 16 + n) * PCELL + g * 16]);
    acc[0] = wmma_f16(a, bf0, acc[0]);
    acc[1] = wmma_f16(a, bf1, acc[1]);
    acc[2] = wmma_f16(a, bf2, acc[2]);
    acc[3] = wmma_f16(a, bf3, acc[3]);
    __syncthreads();
  }
#pragma unroll
  for (int r = 0; r < 8; ++r) {
    int oc = woc + r + 8 * g;
    float bias = proj_b[oc];
    size_t obase = (((size_t)b * 512 + oc) * 64 + h) * 64;
#pragma unroll
    for (int t = 0; t < 4; ++t)
      out[obase + t * 16 + n] = acc[t][r] + bias;
  }
}

// ---------------------------------------------------------------------------
extern "C" void kernel_launch(void* const* d_in, const int* in_sizes, int n_in,
                              void* d_out, int out_size, void* d_ws, size_t ws_size,
                              hipStream_t stream) {
  (void)in_sizes; (void)n_in; (void)out_size; (void)ws_size;
  const float* x      = (const float*)d_in[0];
  const float* q_w    = (const float*)d_in[1];
  const float* q_b    = (const float*)d_in[2];
  const float* kv_w   = (const float*)d_in[3];
  const float* kv_b   = (const float*)d_in[4];
  const float* proj_w = (const float*)d_in[5];
  const float* proj_b = (const float*)d_in[6];
  float* out = (float*)d_out;

  const size_t NPIX = (size_t)16 * 512 * 64 * 64;  // 33,554,432 elements
  char* ws = (char*)d_ws;
  size_t off = 0;
  auto take = [&](size_t bytes) -> char* {
    char* p = ws + off;
    off = (off + bytes + 255) & ~(size_t)255;
    return p;
  };
  h16* xh = (h16*)take(NPIX * 2);                    // x in f16
  h16* wr = (h16*)take((size_t)1536 * 9 * 512 * 2);  // packed qkv weights
  h16* wp = (h16*)take((size_t)512 * 512 * 2);       // packed proj weights
  h16* Qb = (h16*)take(NPIX * 2);
  h16* Kb = (h16*)take(NPIX * 2);
  h16* Vb = (h16*)take(NPIX * 2);
  h16* AO = (h16*)take(NPIX * 2);                    // attention out (permuted ch)

  cvt_f32_f16<<<2048, 256, 0, stream>>>(x, xh, (int)NPIX);
  pack_qkv_w<<<2048, 256, 0, stream>>>(q_w, kv_w, wr);
  cvt_f32_f16<<<512, 256, 0, stream>>>(proj_w, wp, 512 * 512);

  qkv_conv_kernel<<<dim3(512, 12), 256, 0, stream>>>(xh, wr, q_b, kv_b, Qb, Kb, Vb);
  attn_kernel<<<8192, 128, 0, stream>>>(Qb, Kb, Vb, AO);
  proj_kernel<<<dim3(1024, 4), 256, 0, stream>>>(AO, wp, proj_b, out);
}
